// CrossAttention_8486855377137
// MI455X (gfx1250) — compile-verified, matrix operand order checked by reading, not run on hardware
//
#include <hip/hip_runtime.h>

// ---------------------------------------------------------------------------
// Types / helpers
// ---------------------------------------------------------------------------
typedef __attribute__((ext_vector_type(16))) __bf16 v16bf;
typedef __attribute__((ext_vector_type(8)))  __bf16 v8bf;
typedef __attribute__((ext_vector_type(4)))  __bf16 v4bf;
typedef __attribute__((ext_vector_type(8)))  float  v8f;
typedef __attribute__((ext_vector_type(4)))  float  v4f;

#if defined(__AMDGCN__) && __has_builtin(__builtin_amdgcn_tensor_load_to_lds)
#define HAVE_TDM 1
#else
#define HAVE_TDM 0
#endif

__device__ __forceinline__ __bf16 f2bf(float f) {
    union { float f; unsigned u; } v; v.f = f;
    unsigned u = v.u + 0x7FFFu + ((v.u >> 16) & 1u);   // round-nearest-even
    union { unsigned short s; __bf16 b; } o; o.s = (unsigned short)(u >> 16);
    return o.b;
}

__device__ __forceinline__ v16bf mk16(const __bf16* p0, const __bf16* p1) {
    v8bf a = *(const v8bf*)p0;
    v8bf b = *(const v8bf*)p1;
    return __builtin_shufflevector(a, b, 0,1,2,3,4,5,6,7,8,9,10,11,12,13,14,15);
}

__device__ __forceinline__ v8f wmma_bf16(v16bf a, v16bf b, v8f c) {
    return __builtin_amdgcn_wmma_f32_16x16x32_bf16(
        /*neg_a=*/false, a, /*neg_b=*/false, b,
        /*c_mod=*/(short)0, c, /*reuse_a=*/false, /*reuse_b=*/false);
}

// ---------------------------------------------------------------------------
// Problem constants
// ---------------------------------------------------------------------------
#define CB   4
#define CT   2048
#define CS   4096
#define CE   512
#define CKV  2048
#define CH   8
#define CD   64
#define CSCALE 0.125f          // 64^-0.5

// ---------------------------------------------------------------------------
// Tensor Data Mover helpers (D# per CDNA5 ISA)
// ---------------------------------------------------------------------------
#if HAVE_TDM
typedef __attribute__((ext_vector_type(4))) unsigned t_v4u;
typedef __attribute__((ext_vector_type(8))) int      t_v8i;
typedef __attribute__((ext_vector_type(4))) int      t_v4i;

__device__ __forceinline__ unsigned lds_off_of(const void* p) {
    // flat LDS aperture: addr[31:0] is the LDS byte offset
    return (unsigned)(unsigned long long)p;
}

// 2D tensor tile load: data_size = 2 bytes, tile (tile_d0 x tile_d1),
// row stride stride0 (elements), tensor dims for OOB = (tens_d0, tens_d1).
__device__ __forceinline__ void tdm_load_2d(unsigned lds_addr,
                                            unsigned long long gaddr,
                                            int tile_d0, int tile_d1,
                                            int tens_d0, int tens_d1,
                                            int stride0) {
    t_v4u g0;
    g0[0] = 1u;                                            // count=1, user mode
    g0[1] = lds_addr;                                      // lds_addr [63:32]
    g0[2] = (unsigned)gaddr;                               // global_addr lo
    g0[3] = ((unsigned)(gaddr >> 32) & 0x01FFFFFFu)        // global_addr[56:32]
            | 0x80000000u;                                 // type=2 ("image")
    t_v8i g1;
    g1[0] = (1 << 16);                                     // data_size=1 -> 2B
    g1[1] = (tens_d0 & 0xFFFF) << 16;                      // tensor_dim0[15:0]
    g1[2] = ((unsigned)tens_d0 >> 16) | ((tens_d1 & 0xFFFF) << 16);
    g1[3] = ((unsigned)tens_d1 >> 16) | (tile_d0 << 16);   // tile_dim0
    g1[4] = tile_d1;                                       // tile_dim1 (dim2=0)
    g1[5] = stride0;                                       // dim0 stride lo
    g1[6] = 0;                                             // stride hi / dim1 stride
    g1[7] = 0;
    t_v4i z4 = {0, 0, 0, 0};
#if __clang_major__ >= 23
    t_v8i z8 = {0, 0, 0, 0, 0, 0, 0, 0};
    __builtin_amdgcn_tensor_load_to_lds(g0, g1, z4, z4, z8, 0);
#else
    __builtin_amdgcn_tensor_load_to_lds(g0, g1, z4, z4, 0);
#endif
}
#endif // HAVE_TDM

// ---------------------------------------------------------------------------
// f32 -> bf16 conversion (4-wide)
// ---------------------------------------------------------------------------
__global__ void cvt_bf16_kernel(const float* __restrict__ in,
                                __bf16* __restrict__ out, long n) {
    long i = ((long)blockIdx.x * blockDim.x + threadIdx.x) * 4;
    if (i + 3 < n) {
        v4f f = *(const v4f*)(in + i);
        v4bf o;
        o[0] = f2bf(f[0]); o[1] = f2bf(f[1]);
        o[2] = f2bf(f[2]); o[3] = f2bf(f[3]);
        *(v4bf*)(out + i) = o;
    }
}

// ---------------------------------------------------------------------------
// GEMM:  C[M,N] = A[M,K] (bf16) * W[N,K]^T (bf16) + bias[N]
// One wave computes a 16x64 output supertile (A operand reused for 4 WMMAs).
// OUTMODE: 0 = f32 row-major, 1 = bf16 row-major, 2 = bf16 V-transposed
//          ([b][hd][s] with S=4096, used for the V projection)
// ---------------------------------------------------------------------------
template<int OUTMODE>
__global__ __launch_bounds__(256)
void gemm_bf16_kernel(const __bf16* __restrict__ A, const __bf16* __restrict__ W,
                      const float* __restrict__ bias, void* __restrict__ Cout,
                      int M, int N, int K) {
    const int wave = threadIdx.x >> 5;
    const int lane = threadIdx.x & 31;
    const int ln   = lane & 15;
    const int g    = lane >> 4;

    const long tile = (long)blockIdx.x * 8 + wave;   // grid sized exactly
    const int ntn = N >> 6;                          // 64-col supertiles
    const int tm  = (int)(tile / ntn);
    const int tn  = (int)(tile % ntn);

    const __bf16* arow = A + (long)(tm * 16 + ln) * K + 8 * g;
    const __bf16* w0   = W + (long)(tn * 64 + ln) * K + 8 * g;
    const __bf16* w1   = w0 + (long)16 * K;
    const __bf16* w2   = w0 + (long)32 * K;
    const __bf16* w3   = w0 + (long)48 * K;

    v8f acc0 = {}, acc1 = {}, acc2 = {}, acc3 = {};
    for (int k = 0; k < K; k += 32) {
        __builtin_prefetch((const void*)(arow + k + 1024), 0, 1);
        const v16bf av = mk16(arow + k, arow + k + 16);
        acc0 = wmma_bf16(av, mk16(w0 + k, w0 + k + 16), acc0);
        acc1 = wmma_bf16(av, mk16(w1 + k, w1 + k + 16), acc1);
        acc2 = wmma_bf16(av, mk16(w2 + k, w2 + k + 16), acc2);
        acc3 = wmma_bf16(av, mk16(w3 + k, w3 + k + 16), acc3);
    }

#pragma unroll
    for (int j = 0; j < 4; ++j) {
        const v8f acc = (j == 0) ? acc0 : (j == 1) ? acc1 : (j == 2) ? acc2 : acc3;
        const int col = tn * 64 + 16 * j + ln;
        const float bv = bias[col];
#pragma unroll
        for (int r = 0; r < 8; ++r) {
            const int row = tm * 16 + r + 8 * g;     // C/D layout: vgpr r, half g
            const float v = acc[r] + bv;
            if (OUTMODE == 0) {
                ((float*)Cout)[(long)row * N + col] = v;
            } else if (OUTMODE == 1) {
                ((__bf16*)Cout)[(long)row * N + col] = f2bf(v);
            } else {
                const int bb = row >> 12;            // row = b*4096 + s
                const int ss = row & 4095;
                ((__bf16*)Cout)[((long)(bb * CE + col)) * CS + ss] = f2bf(v);
            }
        }
    }
}

// ---------------------------------------------------------------------------
// Flash attention.
// Block = 8 waves, all on the same (b, h); each wave owns one 16-row Q tile.
// K/V chunks (32 S-columns) are staged into double-buffered LDS by the
// Tensor Data Mover (async, s_wait_tensorcnt) and shared by all 8 waves.
// Q layout [b][t][h*64+d] bf16, K layout [b][s][h*64+d] bf16,
// V layout [b][h*64+d][s] bf16 (transposed), O layout [b][t][h*64+d] bf16.
// ---------------------------------------------------------------------------
__global__ __launch_bounds__(256)
void attn_kernel(const __bf16* __restrict__ Q, const __bf16* __restrict__ Km,
                 const __bf16* __restrict__ Vt, __bf16* __restrict__ O) {
    __shared__ __align__(16) __bf16 Kl[2][32][64];   // 2 x 4KB
    __shared__ __align__(16) __bf16 Vl[2][64][32];   // 2 x 4KB
    __shared__ __align__(16) __bf16 Plds[8][16][32]; // 1KB per wave

    const int wave = threadIdx.x >> 5;
    const int lane = threadIdx.x & 31;
    const int ln   = lane & 15;
    const int g    = lane >> 4;
    const int tid  = threadIdx.x;

    const int blk = blockIdx.x;                // 512 blocks = 4b * 8h * 16tc
    const int b   = blk >> 7;
    const int h   = (blk >> 4) & 7;
    const int t0  = ((blk & 15) << 7) + (wave << 4);

    const __bf16* kgbase = Km + ((long)b * CS) * CE + h * CD;
    const __bf16* vgbase = Vt + ((long)(b * CE + h * CD)) * CS;

    // Q tile as two A-operands (k-steps d=0..31 and d=32..63)
    const __bf16* qrow = Q + ((long)(b * CT + t0 + ln)) * CE + h * CD;
    const v16bf qa0 = mk16(qrow +      8 * g, qrow + 16 + 8 * g);
    const v16bf qa1 = mk16(qrow + 32 + 8 * g, qrow + 48 + 8 * g);

    float mrow[8], lrow[8];
    v8f o0 = {}, o1 = {}, o2 = {}, o3 = {};
#pragma unroll
    for (int r = 0; r < 8; ++r) { mrow[r] = -1e30f; lrow[r] = 0.f; }

    auto stage_issue = [&](int buf, int s) {
#if HAVE_TDM
        if (wave == 0) {
            // K chunk: 32 rows (s) x 64 cols (d), row stride 512 elements
            tdm_load_2d(lds_off_of(&Kl[buf][0][0]),
                        (unsigned long long)(const void*)(kgbase + (long)s * CE),
                        /*tile*/64, 32, /*tensor*/64, CS, CE);
            // V chunk: 64 rows (d) x 32 cols (s), row stride 4096 elements
            tdm_load_2d(lds_off_of(&Vl[buf][0][0]),
                        (unsigned long long)(const void*)(vgbase + s),
                        /*tile*/32, 64, /*tensor*/32, 64, CS);
        }
#else
        {   // cooperative fallback: 256 threads x 16B for each of K and V
            const int krow = tid >> 3, kpart = tid & 7;
            *(v8bf*)&Kl[buf][krow][kpart * 8] =
                *(const v8bf*)(kgbase + (long)(s + krow) * CE + kpart * 8);
            const int vrow = tid >> 2, vpart = tid & 3;
            *(v8bf*)&Vl[buf][vrow][vpart * 8] =
                *(const v8bf*)(vgbase + (long)vrow * CS + s + vpart * 8);
        }
#endif
    };
    auto stage_wait = [&]() {
#if HAVE_TDM
#if __has_builtin(__builtin_amdgcn_s_wait_tensorcnt)
        if (wave == 0) __builtin_amdgcn_s_wait_tensorcnt(0);
#endif
#endif
        __syncthreads();
    };

    stage_issue(0, 0);
    stage_wait();

    for (int s = 0; s < CS; s += 32) {
        const int cur = (s >> 5) & 1;
        if (s + 32 < CS) stage_issue(cur ^ 1, s + 32);   // overlap with compute

        // ---- scores: S0 = cols s..s+15, S1 = cols s+16..s+31 (from LDS) ----
        v8f s0 = {}, s1 = {};
        s0 = wmma_bf16(qa0, mk16(&Kl[cur][ln][8*g],         &Kl[cur][ln][16 + 8*g]), s0);
        s0 = wmma_bf16(qa1, mk16(&Kl[cur][ln][32 + 8*g],    &Kl[cur][ln][48 + 8*g]), s0);
        s1 = wmma_bf16(qa0, mk16(&Kl[cur][16+ln][8*g],      &Kl[cur][16+ln][16 + 8*g]), s1);
        s1 = wmma_bf16(qa1, mk16(&Kl[cur][16+ln][32 + 8*g], &Kl[cur][16+ln][48 + 8*g]), s1);

        // ---- online softmax update (row m = r + 8g lives in 16 lanes) ----
        float p0[8], p1[8];
#pragma unroll
        for (int r = 0; r < 8; ++r) {
            float a = s0[r] * CSCALE;
            float c = s1[r] * CSCALE;
            float mx = fmaxf(a, c);
            mx = fmaxf(mx, __shfl_xor(mx, 1));
            mx = fmaxf(mx, __shfl_xor(mx, 2));
            mx = fmaxf(mx, __shfl_xor(mx, 4));
            mx = fmaxf(mx, __shfl_xor(mx, 8));
            const float mnew  = fmaxf(mrow[r], mx);
            const float alpha = __expf(mrow[r] - mnew);
            const float e0 = __expf(a - mnew);
            const float e1 = __expf(c - mnew);
            float rs = e0 + e1;
            rs += __shfl_xor(rs, 1);
            rs += __shfl_xor(rs, 2);
            rs += __shfl_xor(rs, 4);
            rs += __shfl_xor(rs, 8);
            lrow[r] = lrow[r] * alpha + rs;
            mrow[r] = mnew;
            o0[r] *= alpha; o1[r] *= alpha; o2[r] *= alpha; o3[r] *= alpha;
            p0[r] = e0; p1[r] = e1;
        }

        // ---- re-layout P (C/D layout -> A-operand layout) through LDS ----
        __syncthreads();
#pragma unroll
        for (int r = 0; r < 8; ++r) {
            Plds[wave][r + 8 * g][ln]      = f2bf(p0[r]);
            Plds[wave][r + 8 * g][16 + ln] = f2bf(p1[r]);
        }
        __syncthreads();
        const v16bf pa = mk16(&Plds[wave][ln][8 * g], &Plds[wave][ln][16 + 8 * g]);

        // ---- O += P (16x32) * V (32x64), V chunk in LDS as [d][s] ----
        o0 = wmma_bf16(pa, mk16(&Vl[cur][ln][8*g],      &Vl[cur][ln][16 + 8*g]), o0);
        o1 = wmma_bf16(pa, mk16(&Vl[cur][16+ln][8*g],   &Vl[cur][16+ln][16 + 8*g]), o1);
        o2 = wmma_bf16(pa, mk16(&Vl[cur][32+ln][8*g],   &Vl[cur][32+ln][16 + 8*g]), o2);
        o3 = wmma_bf16(pa, mk16(&Vl[cur][48+ln][8*g],   &Vl[cur][48+ln][16 + 8*g]), o3);

        __syncthreads();                     // everyone done reading buffer cur
        if (s + 32 < CS) stage_wait();       // next buffer landed
    }

    // ---- epilogue: O /= l, store bf16 [b][t][h*64+d] ----
#pragma unroll
    for (int r = 0; r < 8; ++r) {
        const float inv = 1.0f / lrow[r];
        __bf16* orow = O + ((long)(b * CT + t0 + r + 8 * g)) * CE + h * CD + ln;
        orow[0]  = f2bf(o0[r] * inv);
        orow[16] = f2bf(o1[r] * inv);
        orow[32] = f2bf(o2[r] * inv);
        orow[48] = f2bf(o3[r] * inv);
    }
}

// ---------------------------------------------------------------------------
// Host launcher
// ---------------------------------------------------------------------------
extern "C" void kernel_launch(void* const* d_in, const int* in_sizes, int n_in,
                              void* d_out, int out_size, void* d_ws, size_t ws_size,
                              hipStream_t stream) {
    const float* x     = (const float*)d_in[0];
    const float* ctx   = (const float*)d_in[1];
    const float* wq_w  = (const float*)d_in[2];
    const float* wq_b  = (const float*)d_in[3];
    const float* wk_w  = (const float*)d_in[4];
    const float* wk_b  = (const float*)d_in[5];
    const float* wv_w  = (const float*)d_in[6];
    const float* wv_b  = (const float*)d_in[7];
    const float* wo_w  = (const float*)d_in[8];
    const float* wo_b  = (const float*)d_in[9];
    float* out = (float*)d_out;

    char* ws = (char*)d_ws;
    size_t off = 0;
    auto carve = [&](size_t bytes) -> void* {
        off = (off + 255) & ~(size_t)255;
        void* p = ws + off;
        off += bytes;
        return p;
    };
    const long nx = (long)CB * CT * CE;
    const long nc = (long)CB * CS * CKV;
    const long nq = (long)CE * CE;
    const long nk = (long)CE * CKV;
    __bf16* xb  = (__bf16*)carve(nx * 2);
    __bf16* cb  = (__bf16*)carve(nc * 2);
    __bf16* wqb = (__bf16*)carve(nq * 2);
    __bf16* wkb = (__bf16*)carve(nk * 2);
    __bf16* wvb = (__bf16*)carve(nk * 2);
    __bf16* wob = (__bf16*)carve(nq * 2);
    __bf16* Qb  = (__bf16*)carve((long)CB * CT * CE * 2);
    __bf16* Kb  = (__bf16*)carve((long)CB * CS * CE * 2);
    __bf16* Vtw = (__bf16*)carve((long)CB * CS * CE * 2);
    __bf16* Ob  = (__bf16*)carve((long)CB * CT * CE * 2);

    auto cvt = [&](const float* src, __bf16* dst, long n) {
        long blocks = (n / 4 + 255) / 256;
        cvt_bf16_kernel<<<(unsigned)blocks, 256, 0, stream>>>(src, dst, n);
    };
    cvt(x,    xb,  nx);
    cvt(ctx,  cb,  nc);
    cvt(wq_w, wqb, nq);
    cvt(wk_w, wkb, nk);
    cvt(wv_w, wvb, nk);
    cvt(wo_w, wob, nq);

    // Q projection: (8192 x 512) -> bf16 row-major. 4096 waves / 8 = 512 blocks
    gemm_bf16_kernel<1><<<512, 256, 0, stream>>>(xb, wqb, wq_b, Qb,
                                                 CB * CT, CE, CE);
    // K projection: (16384 x 512), K=2048. 8192 waves -> 1024 blocks
    gemm_bf16_kernel<1><<<1024, 256, 0, stream>>>(cb, wkb, wk_b, Kb,
                                                  CB * CS, CE, CKV);
    // V projection: stored transposed [b][h*64+d][s]
    gemm_bf16_kernel<2><<<1024, 256, 0, stream>>>(cb, wvb, wv_b, Vtw,
                                                  CB * CS, CE, CKV);
    // Flash attention: 512 blocks * 8 waves (one block per (b,h,128-row chunk))
    attn_kernel<<<512, 256, 0, stream>>>(Qb, Kb, Vtw, Ob);
    // Output projection -> f32 d_out
    gemm_bf16_kernel<0><<<512, 256, 0, stream>>>(Ob, wob, wo_b, out,
                                                 CB * CT, CE, CE);
}
